// Qwen3MoeSparseMoeBlock_39548058861859
// MI455X (gfx1250) — compile-verified, matrix operand order checked by reading, not run on hardware
//
#include <hip/hip_runtime.h>
#include <hip/hip_bf16.h>

#define E   64
#define KK  8
#define H   2048
#define I   768
#define CAP 512
#define T   2048
#define BK  32
#define LDT 40   // BK + 8 ushort pad (80B rows, 16B aligned)

typedef __attribute__((ext_vector_type(16))) __bf16 v16bf;
typedef __attribute__((ext_vector_type(8)))  float  v8f;

struct U128 { unsigned int a, b, c, d; };
struct U64  { unsigned int lo, hi; };

// Native RNE fp32 -> bf16 (backend picks the hw convert path on gfx1250).
__device__ __forceinline__ unsigned short f2bf(float f) {
    __bf16 h = (__bf16)f;
    return __builtin_bit_cast(unsigned short, h);
}
__device__ __forceinline__ unsigned pack2(float a, float b) {
    return (unsigned)f2bf(a) | ((unsigned)f2bf(b) << 16);
}

__device__ __forceinline__ v8f vzero8() {
    v8f z;
    #pragma unroll
    for (int i = 0; i < 8; ++i) z[i] = 0.0f;
    return z;
}

// CDNA5 async copy: 16B global -> LDS, tracked by ASYNCcnt (no VGPR round-trip).
// Generic pointer low 32 bits == LDS byte offset (flat-LDS aperture truncation).
__device__ __forceinline__ void async_copy16(const void* gp, void* lp) {
    unsigned lds = (unsigned)(unsigned long long)lp;
    asm volatile("global_load_async_to_lds_b128 %0, %1, off"
                 :: "v"(lds), "v"(gp)
                 : "memory");
}
__device__ __forceinline__ void async_wait0() {
    asm volatile("s_wait_asynccnt 0x0" ::: "memory");
}

// Load one 16-bit A/B WMMA fragment from an LDS tile stored row-major with
// stride LDT ushorts. Per ISA 16-bit A 16x32 layout: lane (hi = lane>>4) holds
// two contiguous 8-element chunks: [hi*8 .. hi*8+7] and [16+hi*8 .. 16+hi*8+7].
__device__ __forceinline__ v16bf load_frag(const unsigned short* lds, int row, int hi) {
    union { U128 q[2]; v16bf v; } f;
    const unsigned short* p = lds + row * LDT;
    f.q[0] = *(const U128*)(p + hi * 8);
    f.q[1] = *(const U128*)(p + 16 + hi * 8);
    return f.v;
}

// ---------------------------------------------------------------- init -----
__global__ __launch_bounds__(256) void k_init(float* __restrict__ y,
                                              int* __restrict__ rowmap,
                                              float* __restrict__ wmap) {
    int i = blockIdx.x * 256 + threadIdx.x;
    if (i < T * H) y[i] = 0.0f;
    if (i < E * CAP) { rowmap[i] = -1; wmap[i] = 0.0f; }
}

// ---------------------------------------------------------------- router ---
__global__ __launch_bounds__(64) void k_gate(const float* __restrict__ x,
                                             const float* __restrict__ gw,
                                             int* __restrict__ topk_id,
                                             float* __restrict__ topk_w) {
    __shared__ float xs[H];
    __shared__ float lg[E];
    int t = blockIdx.x, tid = threadIdx.x;
    for (int i = tid; i < H; i += 64) xs[i] = x[(size_t)t * H + i];
    __syncthreads();
    const float* g = gw + (size_t)tid * H;
    float acc = 0.0f;
    for (int h = 0; h < H; ++h) acc += xs[h] * g[h];
    lg[tid] = acc;
    __syncthreads();
    if (tid == 0) {
        float mx = lg[0];
        for (int i = 1; i < E; ++i) mx = lg[i] > mx ? lg[i] : mx;
        float p[E];
        for (int i = 0; i < E; ++i) p[i] = __expf(lg[i] - mx);
        float wsum = 0.0f; int ids[KK]; float ws[KK];
        for (int k = 0; k < KK; ++k) {
            int best = 0; float bv = -1.0f;
            for (int i = 0; i < E; ++i) if (p[i] > bv) { bv = p[i]; best = i; }
            ids[k] = best; ws[k] = bv; wsum += bv; p[best] = -2.0f;
        }
        float inv = 1.0f / wsum;
        for (int k = 0; k < KK; ++k) {
            topk_id[t * KK + k] = ids[k];
            topk_w[t * KK + k] = ws[k] * inv;
        }
    }
}

// ---------------------------------------------------- slot assignment ------
// One block per expert; ordered scan over all T*K pairs reproduces the
// reference's cumsum-based capacity positions deterministically.
__global__ __launch_bounds__(256) void k_slots(const int* __restrict__ topk_id,
                                               const float* __restrict__ topk_w,
                                               int* __restrict__ rowmap,
                                               float* __restrict__ wmap,
                                               int* __restrict__ counts) {
    int e = blockIdx.x, tid = threadIdx.x;
    __shared__ int sf[256];
    int base = 0;
    for (int chunk = 0; chunk < T * KK; chunk += 256) {
        int p = chunk + tid;
        int f = (topk_id[p] == e) ? 1 : 0;
        sf[tid] = f;
        __syncthreads();
        for (int off = 1; off < 256; off <<= 1) {           // inclusive scan
            int v = (tid >= off) ? sf[tid - off] : 0;
            __syncthreads();
            sf[tid] += v;
            __syncthreads();
        }
        if (f) {
            int slot = base + sf[tid] - 1;
            if (slot < CAP) {
                rowmap[e * CAP + slot] = p / KK;
                wmap[e * CAP + slot]   = topk_w[p];
            }
        }
        base += sf[255];
        __syncthreads();
    }
    if (tid == 0) counts[e] = base < CAP ? base : CAP;
}

// ----------------------------------------------- fused gate/up GEMM --------
// Per expert: h = silu(X@wg) * (X@wu);  X rows gathered via rowmap,
// fp32 -> bf16 on the fly, bf16 WMMA, bf16 output to Hbuf.
#define GU_BM 64
#define GU_BN 128
__global__ __launch_bounds__(256) void k_gateup(const float* __restrict__ x,
                                                const float* __restrict__ wg,
                                                const float* __restrict__ wu,
                                                const int* __restrict__ rowmap,
                                                const int* __restrict__ counts,
                                                unsigned short* __restrict__ Hbuf) {
    int e = blockIdx.y;
    int cnt = counts[e];
    int tm = blockIdx.x % 8;          // 512/64 = 8 M tiles
    int tn = blockIdx.x / 8;          // 768/128 = 6 N tiles
    if (tm * GU_BM >= cnt) return;    // skip empty capacity tiles

    __shared__ unsigned short As[GU_BM * LDT];
    __shared__ unsigned short BgT[GU_BN * LDT];
    __shared__ unsigned short BuT[GU_BN * LDT];
    __shared__ int rowTok[GU_BM];

    int tid = threadIdx.x;
    int lane = tid & 31, wave = tid >> 5;
    int wm = wave >> 2, wn = wave & 3;      // 2 x 4 waves, 32x32 per wave
    int nl = lane & 15, hi = lane >> 4;

    if (tid < GU_BM) rowTok[tid] = rowmap[e * CAP + tm * GU_BM + tid];
    __syncthreads();

    v8f accG[2][2], accU[2][2];
    #pragma unroll
    for (int r = 0; r < 2; ++r)
        #pragma unroll
        for (int c = 0; c < 2; ++c) { accG[r][c] = vzero8(); accU[r][c] = vzero8(); }

    const float* wgE = wg + (size_t)e * H * I;
    const float* wuE = wu + (size_t)e * H * I;

    for (int k0 = 0; k0 < H; k0 += BK) {
        // stage A (64x32): gather token rows, convert to bf16, packed b64 stores
        #pragma unroll
        for (int it = 0; it < 2; ++it) {
            int idx = it * 256 + tid;         // 512 float4 chunks
            int row = idx >> 3;
            int c4  = (idx & 7) << 2;
            int tok = rowTok[row];
            float4 v = make_float4(0.f, 0.f, 0.f, 0.f);
            if (tok >= 0) v = *(const float4*)(x + (size_t)tok * H + k0 + c4);
            U64 pk; pk.lo = pack2(v.x, v.y); pk.hi = pack2(v.z, v.w);
            *(U64*)&As[row * LDT + c4] = pk;
        }
        // stage Bg/Bu (32x128) transposed -> [n][k], convert to bf16
        #pragma unroll
        for (int it = 0; it < 4; ++it) {
            int idx = it * 256 + tid;         // 1024 float4 chunks
            int kk  = idx >> 5;
            int n4  = (idx & 31) << 2;
            size_t off = (size_t)(k0 + kk) * I + tn * GU_BN + n4;
            float4 vg = *(const float4*)(wgE + off);
            float4 vu = *(const float4*)(wuE + off);
            BgT[(n4 + 0) * LDT + kk] = f2bf(vg.x);
            BgT[(n4 + 1) * LDT + kk] = f2bf(vg.y);
            BgT[(n4 + 2) * LDT + kk] = f2bf(vg.z);
            BgT[(n4 + 3) * LDT + kk] = f2bf(vg.w);
            BuT[(n4 + 0) * LDT + kk] = f2bf(vu.x);
            BuT[(n4 + 1) * LDT + kk] = f2bf(vu.y);
            BuT[(n4 + 2) * LDT + kk] = f2bf(vu.z);
            BuT[(n4 + 3) * LDT + kk] = f2bf(vu.w);
        }
        __syncthreads();

        v16bf af[2];
        #pragma unroll
        for (int r = 0; r < 2; ++r) af[r] = load_frag(As, wm * 32 + r * 16 + nl, hi);
        #pragma unroll
        for (int c = 0; c < 2; ++c) {
            v16bf bg = load_frag(BgT, wn * 32 + c * 16 + nl, hi);
            v16bf bu = load_frag(BuT, wn * 32 + c * 16 + nl, hi);
            #pragma unroll
            for (int r = 0; r < 2; ++r) {
                accG[r][c] = __builtin_amdgcn_wmma_f32_16x16x32_bf16(
                    false, af[r], false, bg, (short)0, accG[r][c], false, false);
                accU[r][c] = __builtin_amdgcn_wmma_f32_16x16x32_bf16(
                    false, af[r], false, bu, (short)0, accU[r][c], false, false);
            }
        }
        __syncthreads();
    }

    // epilogue: silu(g)*u -> bf16
    #pragma unroll
    for (int r = 0; r < 2; ++r)
        #pragma unroll
        for (int c = 0; c < 2; ++c)
            #pragma unroll
            for (int v = 0; v < 8; ++v) {
                int m = tm * GU_BM + wm * 32 + r * 16 + hi * 8 + v;
                int n = tn * GU_BN + wn * 32 + c * 16 + nl;
                float g = accG[r][c][v], u = accU[r][c][v];
                float s = g / (1.0f + __expf(-g));
                Hbuf[((size_t)e * CAP + m) * I + n] = f2bf(s * u);
            }
}

// ----------------------------------------------- down GEMM + scatter -------
#define D_BM 128
#define D_BN 128
__global__ __launch_bounds__(256) void k_down(const unsigned short* __restrict__ Hbuf,
                                              const float* __restrict__ wd,
                                              const int* __restrict__ rowmap,
                                              const float* __restrict__ wmap,
                                              const int* __restrict__ counts,
                                              float* __restrict__ y) {
    int e = blockIdx.y;
    int cnt = counts[e];
    int tm = blockIdx.x & 3;          // 512/128 = 4 M tiles
    int tn = blockIdx.x >> 2;         // 2048/128 = 16 N tiles
    if (tm * D_BM >= cnt) return;

    __shared__ unsigned short As[D_BM * LDT];
    __shared__ unsigned short BdT[D_BN * LDT];
    __shared__ int   rowTok[D_BM];
    __shared__ float rowW[D_BM];

    int tid = threadIdx.x;
    int lane = tid & 31, wave = tid >> 5;
    int wm = wave >> 2, wn = wave & 3;      // 2 x 4 waves, 64x32 per wave
    int nl = lane & 15, hi = lane >> 4;

    if (tid < D_BM) {
        rowTok[tid] = rowmap[e * CAP + tm * D_BM + tid];
        rowW[tid]   = wmap[e * CAP + tm * D_BM + tid];
    }
    __syncthreads();

    v8f acc[4][2];
    #pragma unroll
    for (int r = 0; r < 4; ++r)
        #pragma unroll
        for (int c = 0; c < 2; ++c) acc[r][c] = vzero8();

    const unsigned short* hE = Hbuf + (size_t)e * CAP * I;
    const float* wdE = wd + (size_t)e * I * H;

    for (int k0 = 0; k0 < I; k0 += BK) {
        // stage A (128x32) bf16: async 16B copies global -> LDS (ASYNCcnt)
        #pragma unroll
        for (int it = 0; it < 2; ++it) {
            int idx = it * 256 + tid;         // 512 chunks of 8 bf16
            int row = idx >> 2;
            int c8  = (idx & 3) << 3;
            async_copy16(hE + (size_t)(tm * D_BM + row) * I + k0 + c8,
                         &As[row * LDT + c8]);
        }
        // stage B (32x128) fp32 -> bf16 transposed
        #pragma unroll
        for (int it = 0; it < 4; ++it) {
            int idx = it * 256 + tid;
            int kk  = idx >> 5;
            int n4  = (idx & 31) << 2;
            float4 v = *(const float4*)(wdE + (size_t)(k0 + kk) * H + tn * D_BN + n4);
            BdT[(n4 + 0) * LDT + kk] = f2bf(v.x);
            BdT[(n4 + 1) * LDT + kk] = f2bf(v.y);
            BdT[(n4 + 2) * LDT + kk] = f2bf(v.z);
            BdT[(n4 + 3) * LDT + kk] = f2bf(v.w);
        }
        async_wait0();
        __syncthreads();

        v16bf af[4];
        #pragma unroll
        for (int r = 0; r < 4; ++r) af[r] = load_frag(As, wm * 64 + r * 16 + nl, hi);
        #pragma unroll
        for (int c = 0; c < 2; ++c) {
            v16bf bf_ = load_frag(BdT, wn * 32 + c * 16 + nl, hi);
            #pragma unroll
            for (int r = 0; r < 4; ++r)
                acc[r][c] = __builtin_amdgcn_wmma_f32_16x16x32_bf16(
                    false, af[r], false, bf_, (short)0, acc[r][c], false, false);
        }
        __syncthreads();
    }

    // epilogue: weighted scatter-add into y
    #pragma unroll
    for (int r = 0; r < 4; ++r)
        #pragma unroll
        for (int c = 0; c < 2; ++c)
            #pragma unroll
            for (int v = 0; v < 8; ++v) {
                int ml = wm * 64 + r * 16 + hi * 8 + v;
                int tok = rowTok[ml];
                if (tok < 0) continue;
                int n = tn * D_BN + wn * 32 + c * 16 + nl;
                atomicAdd(&y[(size_t)tok * H + n], acc[r][c][v] * rowW[ml]);
            }
}

// ---------------------------------------------------------------- launch ---
extern "C" void kernel_launch(void* const* d_in, const int* in_sizes, int n_in,
                              void* d_out, int out_size, void* d_ws, size_t ws_size,
                              hipStream_t stream) {
    const float* x  = (const float*)d_in[0];
    const float* gw = (const float*)d_in[1];
    const float* wg = (const float*)d_in[2];
    const float* wu = (const float*)d_in[3];
    const float* wd = (const float*)d_in[4];
    float* y = (float*)d_out;

    char* w = (char*)d_ws;
    int*   topk_id = (int*)w;   w += (size_t)T * KK * sizeof(int);
    float* topk_w  = (float*)w; w += (size_t)T * KK * sizeof(float);
    int*   rowmap  = (int*)w;   w += (size_t)E * CAP * sizeof(int);
    float* wmap    = (float*)w; w += (size_t)E * CAP * sizeof(float);
    int*   counts  = (int*)w;   w += 256;
    unsigned short* Hbuf = (unsigned short*)w;   // E*CAP*I bf16 = 50.3 MB

    k_init  <<<(T * H + 255) / 256, 256, 0, stream>>>(y, rowmap, wmap);
    k_gate  <<<T, 64, 0, stream>>>(x, gw, topk_id, topk_w);
    k_slots <<<E, 256, 0, stream>>>(topk_id, topk_w, rowmap, wmap, counts);
    k_gateup<<<dim3(48, E), 256, 0, stream>>>(x, wg, wu, rowmap, counts, Hbuf);
    k_down  <<<dim3(64, E), 256, 0, stream>>>(Hbuf, wd, rowmap, wmap, counts, y);
}
